// Convblock_84748294685453
// MI455X (gfx1250) — matrix operand order repaired and checked by reading, not compiled, not run on hardware
//
#include <hip/hip_runtime.h>
#include <hip/hip_bf16.h>
#include <math.h>

// ---------------------------------------------------------------------------
// MI455X / gfx1250 fused ConvBlock:
//   GEMM1 (x @ weights -> 448 dyn-weights) via v_wmma_f32_16x16x32_f16,
//   branch-free fast tanh + exact replication of the reference's scrambled
//   im2col gather, skip + LayerNorm, GEMM2 (1x1 conv) via WMMA,
//   bias + PReLU + residual.
// ---------------------------------------------------------------------------

typedef __attribute__((ext_vector_type(16))) _Float16 v16h;
typedef __attribute__((ext_vector_type(8)))  float    v8f;

#define T_DIM   16384
#define C_DIM   64
#define K_TAP   7
#define NOUT    448              // C_DIM * K_TAP
#define ROWS_PB 128              // t-rows per block
#define THREADS 256              // 8 wave32

union HPack { unsigned u; _Float16 h[2]; };
union Frag16 { v16h v; unsigned u[8]; };

// Branch-free tanh: sign(x) * (1-e)/(1+e), e = exp2(-2*log2e*|x|).
// Uses v_exp_f32 / v_rcp_f32 (TRANS pipe, co-executes with WMMA); for large
// |x| e underflows to 0 -> result +-1. No EXEC manipulation.
__device__ inline float fast_tanh(float x) {
    float ax = __builtin_fabsf(x);
    float e  = __builtin_amdgcn_exp2f(ax * -2.8853900817779268f); // -2/ln2^-1
    float r  = (1.0f - e) * __builtin_amdgcn_rcpf(1.0f + e);
    return __builtin_copysignf(r, x);
}

// ---------------------------------------------------------------------------
// Prep: pack weights (C,C,K) and conv_kernel (C,C) as f16, K-pair-major so a
// WMMA B fragment VGPR is one contiguous b32 load.
//   Bh2[k2*448 + j] = pack(f16 W[d, 2k2, kw], f16 W[d, 2k2+1, kw]),  j = d*7+kw
//   Ck2[k2*64  + n] = pack(f16 CK[2k2, n],    f16 CK[2k2+1, n])
// ---------------------------------------------------------------------------
__global__ void convblock_prep(const float* __restrict__ weights,
                               const float* __restrict__ ck,
                               unsigned* __restrict__ Bh2,
                               unsigned* __restrict__ Ck2) {
    int tid = blockIdx.x * blockDim.x + threadIdx.x;
    int stride = gridDim.x * blockDim.x;
    for (int i = tid; i < 32 * NOUT; i += stride) {
        int k2 = i / NOUT, j = i - k2 * NOUT;
        int d = j / K_TAP, kw = j - d * K_TAP;
        HPack p;
        p.h[0] = (_Float16)weights[d * (C_DIM * K_TAP) + (2 * k2)     * K_TAP + kw];
        p.h[1] = (_Float16)weights[d * (C_DIM * K_TAP) + (2 * k2 + 1) * K_TAP + kw];
        Bh2[i] = p.u;
    }
    for (int i = tid; i < 32 * C_DIM; i += stride) {
        int k2 = i / C_DIM, n = i - k2 * C_DIM;
        HPack p;
        p.h[0] = (_Float16)ck[(2 * k2)     * C_DIM + n];
        p.h[1] = (_Float16)ck[(2 * k2 + 1) * C_DIM + n];
        Ck2[i] = p.u;
    }
}

// ---------------------------------------------------------------------------
// Fused main kernel. One block = 128 t-rows (all inside one batch b since
// 128 | 16384); 8 waves, one 16-row WMMA M-tile per wave.
// LDS (48 KB):
//   [0,16K)   Xh  : x tile as f16 [128][64]  -> reused as Ynh after LN
//   [16K,48K) Ys  : f32 y accumulator [128][64] -> reused as Zs (GEMM2 out)
// ---------------------------------------------------------------------------
__global__ __launch_bounds__(THREADS, 1)
void convblock_fused(const float* __restrict__ x,
                     const unsigned* __restrict__ Bh2,
                     const unsigned* __restrict__ Ck2,
                     const float* __restrict__ ln_scale,
                     const float* __restrict__ ln_bias,
                     const float* __restrict__ conv_bias,
                     const float* __restrict__ prelu_slope,
                     float* __restrict__ out) {
    __shared__ __align__(16) unsigned char smem[49152];
    _Float16* Xh  = (_Float16*)smem;                 // [128][64] f16
    float*    Ys  = (float*)(smem + 16384);          // [128][64] f32
    _Float16* Ynh = (_Float16*)smem;                 // region reuse after LN
    float*    Zs  = (float*)(smem + 16384);          // region reuse for z

    const int tid  = threadIdx.x;
    const int lane = tid & 31;
    const int wave = tid >> 5;
    const int m    = lane & 15;        // fragment row/col within half-wave
    const int hsel = lane >> 4;        // which half-wave group

    const int grow = blockIdx.x * ROWS_PB;           // global token row
    const int b    = grow / T_DIM;
    const int t0   = grow - b * T_DIM;
    const float* xb = x + (size_t)b * T_DIM * C_DIM;

    __builtin_prefetch(Bh2, 0, 1);                   // global_prefetch_b8

    // ---- phase 0: stage x tile to f16 LDS, zero y accumulator ------------
    for (int i = tid; i < ROWS_PB * C_DIM; i += THREADS) {
        Xh[i] = (_Float16)xb[(size_t)t0 * C_DIM + i];
        Ys[i] = 0.0f;
    }
    __syncthreads();

    const int row0 = wave * 16;        // this wave's M-tile base (local rows)

    // A fragments for GEMM1 (A = 16x32 f16, contiguous k = 16*hsel + s), 2 k-steps.
    Frag16 a1[2];
#pragma unroll
    for (int ks = 0; ks < 2; ++ks) {
#pragma unroll
        for (int i = 0; i < 8; ++i) {
            int kb = ks * 32 + hsel * 16 + i * 2;
            a1[ks].u[i] = *(const unsigned*)&Xh[(row0 + m) * C_DIM + kb];
        }
    }

    // ---- phase 1: GEMM1 tile-by-tile over N=448, fused tanh + u-gather ---
#pragma unroll 2
    for (int jt = 0; jt < NOUT / 16; ++jt) {
        v8f acc = {};
#pragma unroll
        for (int ks = 0; ks < 2; ++ks) {
            Frag16 bf;                  // B = 32x16 f16: lanes 0-15 K0-15,
#pragma unroll                           // lanes 16-31 K16-31, linear VGPRs
            for (int i = 0; i < 8; ++i) {
                int k2row = ks * 16 + hsel * 8 + i;       // u32 (K-pair) row
                bf.u[i] = Bh2[k2row * NOUT + jt * 16 + m];
            }
            acc = __builtin_amdgcn_wmma_f32_16x16x32_f16(
                false, a1[ks].v, false, bf.v, (short)0, acc, false, false);
        }
        // consume: D[v][lane] = Wdyn[t_row, j]; y[t,d] += tanh(Wdyn)*u
        const int j  = jt * 16 + m;
        const int d  = j / K_TAP;
        const int kw = j - d * K_TAP;
        const int tbase = t0 + row0 + hsel * 8;           // + vv = t_glob
#pragma unroll
        for (int vv = 0; vv < 8; ++vv) {
            int t_glob = tbase + vv;
            // exact decode of the reference's reshape-scrambled im2col:
            int av = 256 * kw + t_glob;
            int q  = av / 448;
            int r  = av - q * 448;
            int kp = r >> 6;
            int sp = r & 63;
            int src    = sp + kp - 3;                 // padded channel index
            int tprime = 256 * d + 36 * kw + q;       // scrambled time index
            // branch-free gather: always load (clamped addr), cndmask result
            bool valid = (unsigned)src < (unsigned)C_DIM;
            int  srcc  = valid ? src : 0;
            float xv   = xb[(size_t)(unsigned)(tprime * C_DIM + srcc)];
            float uval = valid ? xv : 0.0f;
            int t_loc  = row0 + (hsel ? vv + 8 : vv);
            atomicAdd(&Ys[t_loc * C_DIM + d], fast_tanh(acc[vv]) * uval);
        }
    }
    __syncthreads();

    // ---- phase 2: skip add + LayerNorm (2 threads per row, 32 ch each) ---
    const int prow  = tid >> 1;
    const int cbase = (tid & 1) * 32;
    float yn[32];
    float s1 = 0.0f, s2 = 0.0f;
#pragma unroll
    for (int i = 0; i < 32; ++i) {
        int c = cbase + i;
        float v = xb[(size_t)(t0 + prow) * C_DIM + c] + Ys[prow * C_DIM + c];
        yn[i] = v; s1 += v; s2 += v * v;
    }
    s1 += __shfl_xor(s1, 1, 32);       // partner lane holds other 32 channels
    s2 += __shfl_xor(s2, 1, 32);
    float mean = s1 * (1.0f / 64.0f);
    float var  = s2 * (1.0f / 64.0f) - mean * mean;
    float rstd = rsqrtf(var + 1e-6f);
#pragma unroll
    for (int i = 0; i < 32; ++i) {
        int c = cbase + i;
        float v = (yn[i] - mean) * rstd * ln_scale[c] + ln_bias[c];
        yn[i] = v;                                    // keep f32 in regs
        Ynh[prow * C_DIM + c] = (_Float16)v;          // f16 copy for GEMM2 A
    }
    __syncthreads();

    // ---- phase 3: GEMM2 (1x1 conv) via WMMA, bias + PReLU, z -> LDS ------
    const float slope = prelu_slope[0];
    v8f zacc[4] = {};
#pragma unroll
    for (int ks = 0; ks < 2; ++ks) {
        Frag16 af;
#pragma unroll
        for (int i = 0; i < 8; ++i) {
            int kb = ks * 32 + hsel * 16 + i * 2;
            af.u[i] = *(const unsigned*)&Ynh[(row0 + m) * C_DIM + kb];
        }
#pragma unroll
        for (int nt = 0; nt < 4; ++nt) {
            Frag16 bf;
#pragma unroll
            for (int i = 0; i < 8; ++i) {
                int k2row = ks * 16 + hsel * 8 + i;
                bf.u[i] = Ck2[k2row * C_DIM + nt * 16 + m];
            }
            zacc[nt] = __builtin_amdgcn_wmma_f32_16x16x32_f16(
                false, af.v, false, bf.v, (short)0, zacc[nt], false, false);
        }
    }
#pragma unroll
    for (int nt = 0; nt < 4; ++nt) {
#pragma unroll
        for (int vv = 0; vv < 8; ++vv) {
            int rrow = row0 + (hsel ? vv + 8 : vv);
            int col  = nt * 16 + m;
            float z = zacc[nt][vv] + conv_bias[col];
            z = (z >= 0.0f) ? z : slope * z;
            Zs[rrow * C_DIM + col] = z;
        }
    }
    __syncthreads();

    // ---- phase 4: out = yn + z ------------------------------------------
    float* outb = out + (size_t)b * T_DIM * C_DIM;
#pragma unroll
    for (int i = 0; i < 32; ++i) {
        int c = cbase + i;
        outb[(size_t)(t0 + prow) * C_DIM + c] = yn[i] + Zs[prow * C_DIM + c];
    }
}

// ---------------------------------------------------------------------------
extern "C" void kernel_launch(void* const* d_in, const int* in_sizes, int n_in,
                              void* d_out, int out_size, void* d_ws, size_t ws_size,
                              hipStream_t stream) {
    const float* x        = (const float*)d_in[0];
    const float* weights  = (const float*)d_in[1];
    const float* ln_scale = (const float*)d_in[2];
    const float* ln_bias  = (const float*)d_in[3];
    const float* ck       = (const float*)d_in[4];
    const float* cbias    = (const float*)d_in[5];
    const float* slope    = (const float*)d_in[6];

    unsigned* Bh2 = (unsigned*)d_ws;        // 32*448 u32 = 56 KB
    unsigned* Ck2 = Bh2 + 32 * NOUT;        // 32*64  u32 =  8 KB

    convblock_prep<<<28, THREADS, 0, stream>>>(weights, ck, Bh2, Ck2);

    const int blocks = (8 * T_DIM) / ROWS_PB;   // 1024
    convblock_fused<<<blocks, THREADS, 0, stream>>>(
        x, Bh2, Ck2, ln_scale, ln_bias, cbias, slope, (float*)d_out);
}
